// MSDeformAttn_STATrans_v2_51582557225487
// MI455X (gfx1250) — matrix operand order repaired and checked by reading, not compiled, not run on hardware
//
#include <hip/hip_runtime.h>
#include <hip/hip_bf16.h>

// ---------------------------------------------------------------------------
// MS-Deformable-Attention (spatio-temporal variant) for MI455X / gfx1250.
// Dense projections run on v_wmma_f32_16x16x32_bf16 (wave32 WMMA, f32
// accumulate); each wave now computes an NT-wide strip of 16x16 tiles so one
// converted A fragment feeds NT WMMAs per K-step (4x fewer A loads + cvts).
// The bilinear-gather stage reads a 33.5MB f32 `value` buffer that stays
// resident in the 192MB L2; lane==channel makes every corner fetch one
// contiguous 128B line per wave.
// ---------------------------------------------------------------------------

typedef __attribute__((ext_vector_type(16))) __bf16 v16bf;
typedef __attribute__((ext_vector_type(8)))  float  v8f;

#define LV   4      // levels
#define NB   2      // batch
#define QL   4096   // queries per level (64*64)
#define LEN  16384  // LV*QL
#define CM   256    // d_model
#define NH   8
#define NP   4
#define NCOL 96     // 64 offset cols + 32 attn cols

// --------------------------------------------------------------------------
// Pack weights: transpose to bf16 [N][K] so each lane's B fragment is one
// contiguous 32-byte run.  Also fuse W_off|W_attn into one 96-wide matrix
// per level, and pack the matching bias vector.
// --------------------------------------------------------------------------
__global__ void pack_weights(const float* __restrict__ Wval,
                             const float* __restrict__ Woff,
                             const float* __restrict__ boff,
                             const float* __restrict__ Wattn,
                             const float* __restrict__ battn,
                             const float* __restrict__ Wout,
                             __bf16* __restrict__ WvalT,
                             __bf16* __restrict__ WoutT,
                             __bf16* __restrict__ Wp,
                             float*  __restrict__ bias2)
{
    int idx = blockIdx.x * blockDim.x + threadIdx.x;
    if (idx < 65536) {                                   // WvalT[n][k] = Wval[k][n]
        int nn = idx >> 8, k = idx & 255;
        WvalT[idx] = (__bf16)Wval[k * 256 + nn];
    } else if (idx < 131072) {                           // WoutT
        int t = idx - 65536;
        int nn = t >> 8, k = t & 255;
        WoutT[t] = (__bf16)Wout[k * 256 + nn];
    } else if (idx < 131072 + LV * NCOL * 256) {         // Wp[l][n][k]
        int t  = idx - 131072;
        int k  = t & 255;
        int t2 = t >> 8;
        int nn = t2 % NCOL;
        int l  = t2 / NCOL;
        float w = (nn < 64) ? Woff [(l * 256 + k) * 64 + nn]
                            : Wattn[(l * 256 + k) * 32 + (nn - 64)];
        Wp[t] = (__bf16)w;
    } else if (idx < 131072 + LV * NCOL * 256 + LV * NCOL) {
        int t  = idx - (131072 + LV * NCOL * 256);
        int nn = t % NCOL;
        int l  = t / NCOL;
        bias2[t] = (nn < 64) ? boff[l * 64 + nn] : battn[l * 32 + (nn - 64)];
    }
}

// --------------------------------------------------------------------------
// bf16-WMMA GEMM:  C[M][N] = A[M][K](f32) * Bp[N][K](bf16)^T + bias.
// One wave computes a 16 x (16*NT) strip: the A fragment (two 32B f32 loads
// + cvt to bf16) is reused by NT WMMAs per 32-deep K step.
// tilesPerLevel>0 selects per-level B/bias (fused offset+attn projection).
// --------------------------------------------------------------------------
template<int NT>
__global__ void gemm_bf16_wmma(const float*  __restrict__ A,
                               const __bf16* __restrict__ Bp,
                               const float*  __restrict__ bias,
                               float*        __restrict__ C,
                               int M, int N, int K,
                               int tilesPerLevel, int levStrideB, int levStrideBias)
{
    int wid  = (blockIdx.x * blockDim.x + threadIdx.x) >> 5;
    int lane = threadIdx.x & 31;
    int nGroups = N / (16 * NT);
    int mTile   = wid / nGroups;
    int nGroup  = wid - mTile * nGroups;
    if (mTile * 16 >= M) return;

    if (tilesPerLevel > 0) {
        int lev = mTile / tilesPerLevel;
        Bp   += (size_t)lev * levStrideB;
        bias += lev * levStrideBias;
    }

    int r  = lane & 15;     // row (A) / col (B,C) inside tile
    int kg = lane >> 4;     // half-wave K group

    // A fragment: lanes 0-15 carry K 0..15, lanes 16-31 K 16..31
    const float*  arow  = A  + (size_t)(mTile * 16 + r) * K + 16 * kg;
    // B fragments: col = r (+16*t), 16 contiguous K values starting at 16*kg
    const __bf16* brow0 = Bp + (size_t)(nGroup * NT * 16 + r) * K + 16 * kg;

    v8f acc[NT] = {};
#pragma unroll
    for (int kk = 0; kk < 256; kk += 32) {
        v8f a0 = *(const v8f*)(arow + kk);        // K_local = 16*kg     + 0..7
        v8f a1 = *(const v8f*)(arow + kk + 8);    // K_local = 16*kg + 8 + 0..7
        v16bf a;
#pragma unroll
        for (int e = 0; e < 8; ++e) {
            a[e]     = (__bf16)a0[e];
            a[8 + e] = (__bf16)a1[e];
        }
#pragma unroll
        for (int t = 0; t < NT; ++t) {
            v16bf b = *(const v16bf*)(brow0 + (size_t)t * 16 * K + kk);
            acc[t] = __builtin_amdgcn_wmma_f32_16x16x32_bf16(
                         false, a, false, b, (short)0, acc[t], false, false);
        }
    }

#pragma unroll
    for (int t = 0; t < NT; ++t) {
        int col = nGroup * NT * 16 + t * 16 + r;
        float bv = bias[col];
#pragma unroll
        for (int v = 0; v < 8; ++v) {
            int row = mTile * 16 + v + 8 * kg;    // C/D VGPR layout
            C[(size_t)row * N + col] = acc[t][v] + bv;
        }
    }
}

// --------------------------------------------------------------------------
// Sampling stage: one wave per (n, len, h); lane = channel d (D == 32 ==
// wave size).  Softmax over 16 (level,point) logits, bilinear gather from
// L2-resident value, weighted accumulate -> tmp[n][len][h*32+d].
// --------------------------------------------------------------------------
__global__ void msda_sample(const float* __restrict__ value,    // [NB][LEN][256]
                            const float* __restrict__ offattn,  // [LV][LV][NB][QL][96]
                            const float* __restrict__ refpts,   // [NB][LEN][LV][2]
                            const int*   __restrict__ sshapes,  // [LV][2]
                            const int*   __restrict__ lstart,   // [LV]
                            float*       __restrict__ tmp)      // [NB][LEN][256]
{
    int wid  = (blockIdx.x * blockDim.x + threadIdx.x) >> 5;
    int lane = threadIdx.x & 31;
    int h   = wid & (NH - 1);
    int t   = wid >> 3;
    int len = t & (LEN - 1);
    int n   = t >> 14;
    if (n >= NB) return;
    int i = len >> 12;       // temporal level of the query
    int q = len & (QL - 1);

    // ---- softmax over the 16 attention logits (uniform across lanes) ----
    float logits[16];
    float mx = -1e30f;
#pragma unroll
    for (int j = 0; j < LV; ++j) {
        const float* base = offattn + (size_t)(((i * LV + j) * NB + n) * QL + q) * NCOL;
#pragma unroll
        for (int p = 0; p < NP; ++p) {
            float lg = base[64 + h * NP + p];
            logits[j * NP + p] = lg;
            mx = fmaxf(mx, lg);
        }
    }
    float ssum = 0.f;
#pragma unroll
    for (int k = 0; k < 16; ++k) { logits[k] = expf(logits[k] - mx); ssum += logits[k]; }
    float inv = 1.0f / ssum;

    // ---- bilinear sampling, lane d accumulates channel d ----
    float acc = 0.f;
    for (int j = 0; j < LV; ++j) {
        const float* base = offattn + (size_t)(((i * LV + j) * NB + n) * QL + q) * NCOL;
        int Hl = sshapes[2 * j + 0];
        int Wl = sshapes[2 * j + 1];
        int ls = lstart[j];
        float rx = refpts[(((size_t)n * LEN + len) * LV + j) * 2 + 0];
        float ry = refpts[(((size_t)n * LEN + len) * LV + j) * 2 + 1];
        const float* vbase = value + ((size_t)n * LEN + ls) * CM + h * 32 + lane;
#pragma unroll
        for (int p = 0; p < NP; ++p) {
            float ox = base[(h * NP + p) * 2 + 0];
            float oy = base[(h * NP + p) * 2 + 1];
            float px = (rx + ox / (float)Wl) * (float)Wl - 0.5f;
            float py = (ry + oy / (float)Hl) * (float)Hl - 0.5f;
            float fx = floorf(px), fy = floorf(py);
            float dx = px - fx,    dy = py - fy;
            int x0 = (int)fx, y0 = (int)fy;
            float w = logits[j * NP + p] * inv;
            float s = 0.f;
            {   // (x0, y0)
                if (x0 >= 0 && x0 < Wl && y0 >= 0 && y0 < Hl)
                    s += (1.f - dx) * (1.f - dy) * vbase[(size_t)(y0 * Wl + x0) * CM];
            }
            {   // (x0+1, y0)
                int cx = x0 + 1;
                if (cx >= 0 && cx < Wl && y0 >= 0 && y0 < Hl)
                    s += dx * (1.f - dy) * vbase[(size_t)(y0 * Wl + cx) * CM];
            }
            {   // (x0, y0+1)
                int cy = y0 + 1;
                if (x0 >= 0 && x0 < Wl && cy >= 0 && cy < Hl)
                    s += (1.f - dx) * dy * vbase[(size_t)(cy * Wl + x0) * CM];
            }
            {   // (x0+1, y0+1)
                int cx = x0 + 1, cy = y0 + 1;
                if (cx >= 0 && cx < Wl && cy >= 0 && cy < Hl)
                    s += dx * dy * vbase[(size_t)(cy * Wl + cx) * CM];
            }
            acc += w * s;
        }
    }
    tmp[((size_t)n * LEN + len) * CM + h * 32 + lane] = acc;
}

// --------------------------------------------------------------------------
extern "C" void kernel_launch(void* const* d_in, const int* in_sizes, int n_in,
                              void* d_out, int out_size, void* d_ws, size_t ws_size,
                              hipStream_t stream) {
    (void)in_sizes; (void)n_in; (void)out_size; (void)ws_size;

    const float* seq_query = (const float*)d_in[0];   // [4][4][2][4096][256]
    const float* refpts    = (const float*)d_in[1];   // [2][16384][4][2]
    const float* inflat    = (const float*)d_in[2];   // [2][16384][256]
    const float* W_off     = (const float*)d_in[3];   // [4][256][64]
    const float* b_off     = (const float*)d_in[4];   // [4][64]
    const float* W_attn    = (const float*)d_in[5];   // [4][256][32]
    const float* b_attn    = (const float*)d_in[6];   // [4][32]
    const float* W_val     = (const float*)d_in[7];   // [256][256]
    const float* b_val     = (const float*)d_in[8];   // [256]
    const float* W_out     = (const float*)d_in[9];   // [256][256]
    const float* b_out     = (const float*)d_in[10];  // [256]
    const int*   sshapes   = (const int*)d_in[11];    // [4][2]
    const int*   lstart    = (const int*)d_in[12];    // [4]

    // ---- workspace layout (all offsets 256B-aligned), total ~118 MB ----
    char* ws = (char*)d_ws;
    float*  value   = (float*) (ws);                       // 33,554,432 B
    float*  offattn = (float*) (ws + 33554432);            // 50,331,648 B
    float*  tmp     = (float*) (ws + 33554432 + 50331648); // 33,554,432 B
    size_t  wbase   = 33554432 + 50331648 + 33554432;
    __bf16* WvalT   = (__bf16*)(ws + wbase);               // 131,072 B
    __bf16* WoutT   = (__bf16*)(ws + wbase + 131072);      // 131,072 B
    __bf16* Wp      = (__bf16*)(ws + wbase + 262144);      // 196,608 B
    float*  bias2   = (float*) (ws + wbase + 262144 + 196608);

    // 1) pack/transposed bf16 weights + fused bias
    {
        int total = 131072 + LV * NCOL * 256 + LV * NCOL;  // 229,760
        pack_weights<<<(total + 255) / 256, 256, 0, stream>>>(
            W_val, W_off, b_off, W_attn, b_attn, W_out,
            WvalT, WoutT, Wp, bias2);
    }

    // 2) value projection: [32768 x 256] * [256 x 256]
    //    2048 M-tiles x 4 N-groups (16x64 per wave) = 8192 waves
    gemm_bf16_wmma<4><<<1024, 256, 0, stream>>>(
        inflat, WvalT, b_val, value,
        NB * LEN, CM, CM, 0, 0, 0);

    // 3) fused offset+attn projection: [131072 x 256] * [256 x 96] per level
    //    8192 M-tiles x 2 N-groups (16x48 per wave) = 16384 waves
    gemm_bf16_wmma<3><<<2048, 256, 0, stream>>>(
        seq_query, Wp, bias2, offattn,
        LV * LV * NB * QL, NCOL, CM,
        /*tilesPerLevel=*/2048, /*levStrideB=*/NCOL * CM, /*levStrideBias=*/NCOL);

    // 4) softmax + bilinear sampling + head aggregation
    msda_sample<<<32768, 256, 0, stream>>>(
        value, offattn, refpts, sshapes, lstart, tmp);

    // 5) output projection: [32768 x 256] * [256 x 256] -> d_out
    gemm_bf16_wmma<4><<<1024, 256, 0, stream>>>(
        tmp, WoutT, b_out, (float*)d_out,
        NB * LEN, CM, CM, 0, 0, 0);
}
